// GraphConvLayer_53618371723424
// MI455X (gfx1250) — matrix operand order repaired and checked by reading, not compiled, not run on hardware
//
#include <hip/hip_runtime.h>
#include <math.h>

#define DIMS    128
#define NNODES  10000
#define NEDGES  640000
#define BN_EPS  1e-3f

typedef __bf16 bf16_t;
typedef bf16_t v16bf __attribute__((ext_vector_type(16)));
typedef float  v8f   __attribute__((ext_vector_type(8)));

// Branch-free tanh-form GELU:  gelu(x) = x * sigmoid(2u),
// u = 0.7978845608 * x * (1 + 0.044715 x^2).  Max err ~1e-3 (below bf16 noise).
// Lowers to straight-line v_exp_f32 / v_rcp_f32 + FMAs, no divergence.
__device__ __forceinline__ float gelu_fast(float x) {
  const float t = x * x;
  const float z = x * fmaf(t, -0.0713548162f, -1.5957691216f);  // z = -2u
  const float e = __expf(z);                                    // exp(-2u)
  return x * (1.0f / (1.0f + e));                               // x * sigmoid(2u)
}

// A-fragment (16x32 bf16, MxK): lane<16 holds K=[kb..kb+7] and [kb+16..kb+23]
// with kb = (lane>>4)*8. Pointer p is pre-offset to [row][kchunk*32 + kb].
__device__ __forceinline__ v16bf load_fragA(const bf16_t* p) {
  v16bf r;
#pragma unroll
  for (int j = 0; j < 8; ++j) { r[j] = p[j]; r[j + 8] = p[16 + j]; }
  return r;
}

// B-fragment (32x16 bf16, KxN) from LDS weights stored transposed [n][k]:
// lane group holds K=[kb..kb+7] and [kb+16..kb+23] with kb = (lane>>4)*8.
__device__ __forceinline__ v16bf load_fragB(const bf16_t* p) {
  v16bf r;
#pragma unroll
  for (int j = 0; j < 8; ++j) { r[j] = p[j]; r[j + 8] = p[16 + j]; }
  return r;
}

// ---------------------------------------------------------------------------
// zero agg + cnt scratch
// ---------------------------------------------------------------------------
__global__ __launch_bounds__(256) void zero_kernel(float* __restrict__ agg,
                                                   float* __restrict__ cnt) {
  const int nAgg = NNODES * DIMS;
  const int total = nAgg + NNODES;
  for (int i = blockIdx.x * blockDim.x + threadIdx.x; i < total;
       i += gridDim.x * blockDim.x) {
    if (i < nAgg) agg[i] = 0.0f; else cnt[i - nAgg] = 0.0f;
  }
}

// ---------------------------------------------------------------------------
// per-destination edge counts (for segment mean)
// ---------------------------------------------------------------------------
__global__ __launch_bounds__(256) void count_kernel(const int* __restrict__ edges,
                                                    float* __restrict__ cnt) {
  for (int e = blockIdx.x * blockDim.x + threadIdx.x; e < NEDGES;
       e += gridDim.x * blockDim.x) {
    atomicAdd(&cnt[edges[e]], 1.0f);
  }
}

// ---------------------------------------------------------------------------
// Fused edge pipeline:
//   gather neigh -> BN1 -> (bf16) @W1+b1 -> gelu -> BN2 -> @W2+b2 -> gelu
//   -> * edge_weight -> atomic scatter-add into agg
// 256 threads = 8 waves; each wave owns 16 edges of a 128-edge tile.
// ---------------------------------------------------------------------------
__global__ __launch_bounds__(256) void edge_msg_kernel(
    const float* __restrict__ nodeF, const int* __restrict__ edges,
    const float* __restrict__ edgeW,
    const float* __restrict__ p1g, const float* __restrict__ p1b,
    const float* __restrict__ p1m, const float* __restrict__ p1v,
    const float* __restrict__ p1W, const float* __restrict__ p1bias,
    const float* __restrict__ p2g, const float* __restrict__ p2b,
    const float* __restrict__ p2m, const float* __restrict__ p2v,
    const float* __restrict__ p2W, const float* __restrict__ p2bias,
    float* __restrict__ agg) {
  __shared__ bf16_t sWt1[DIMS * DIMS];   // W1 transposed [n][k], bf16
  __shared__ bf16_t sWt2[DIMS * DIMS];   // W2 transposed [n][k], bf16
  __shared__ bf16_t sActA[DIMS * DIMS];  // stage-1 activations (per-wave rows)
  __shared__ bf16_t sActB[DIMS * DIMS];  // stage-2 activations
  __shared__ float sS1[DIMS], sH1[DIMS], sS2[DIMS], sH2[DIMS];
  __shared__ float sB1[DIMS], sB2[DIMS];
  __shared__ float sEw[128];
  __shared__ int   sDst[128];

  const int tid  = threadIdx.x;
  const int wave = tid >> 5;
  const int lane = tid & 31;

  // ---- preamble: weights -> LDS (bf16, transposed), BN folding ----
  for (int i = tid; i < DIMS * DIMS; i += 256) {
    const int k = i >> 7, n = i & (DIMS - 1);
    sWt1[n * DIMS + k] = (bf16_t)p1W[i];
    sWt2[n * DIMS + k] = (bf16_t)p2W[i];
  }
  if (tid < DIMS) {
    const float s1 = p1g[tid] * rsqrtf(p1v[tid] + BN_EPS);
    sS1[tid] = s1; sH1[tid] = p1b[tid] - p1m[tid] * s1;
    const float s2 = p2g[tid] * rsqrtf(p2v[tid] + BN_EPS);
    sS2[tid] = s2; sH2[tid] = p2b[tid] - p2m[tid] * s2;
    sB1[tid] = p1bias[tid]; sB2[tid] = p2bias[tid];
  }
  __syncthreads();

  const int kbA  = (lane >> 4) << 3;
  const int kbB  = (lane >> 4) << 3;
  const int rowA = wave * 16 + (lane & 15);
  const int colB = lane & 15;
  const int rhi  = (lane >> 4) << 3;

  // hoist per-lane loop-invariant BN1 params for the gather columns
  const int cg = lane * 4;
  float s1r[4], h1r[4];
#pragma unroll
  for (int j = 0; j < 4; ++j) { s1r[j] = sS1[cg + j]; h1r[j] = sH1[cg + j]; }

  const int numTiles = NEDGES / 128;
  for (int tile = blockIdx.x; tile < numTiles; tile += gridDim.x) {
    const int base = tile * 128;

    // ---- gather neighbors + BN1, write bf16 tile (wave-local rows) ----
#pragma unroll 4
    for (int r = 0; r < 16; ++r) {
      const int le  = wave * 16 + r;
      const int e   = base + le;
      const int nbr = edges[NEDGES + e];  // edges[1][e]
      const float4 x = *(const float4*)(nodeF + (size_t)nbr * DIMS + cg);
      bf16_t* d = &sActA[le * DIMS + cg];
      d[0] = (bf16_t)fmaf(x.x, s1r[0], h1r[0]);
      d[1] = (bf16_t)fmaf(x.y, s1r[1], h1r[1]);
      d[2] = (bf16_t)fmaf(x.z, s1r[2], h1r[2]);
      d[3] = (bf16_t)fmaf(x.w, s1r[3], h1r[3]);
    }
    if (lane < 16) {
      const int le = wave * 16 + lane;
      sEw[le]  = edgeW[base + le];
      sDst[le] = edges[base + le];      // edges[0][e]
    }

    // ---- stage 1: @W1 + b1, gelu, fold BN2, -> sActB ----
    v16bf aF[4];
#pragma unroll
    for (int kc = 0; kc < 4; ++kc)
      aF[kc] = load_fragA(&sActA[rowA * DIMS + kc * 32 + kbA]);
#pragma unroll
    for (int nt = 0; nt < 8; ++nt) {
      v8f acc = {0.f, 0.f, 0.f, 0.f, 0.f, 0.f, 0.f, 0.f};
#pragma unroll
      for (int kc = 0; kc < 4; ++kc) {
        v16bf bF = load_fragB(&sWt1[(nt * 16 + colB) * DIMS + kc * 32 + kbB]);
        acc = __builtin_amdgcn_wmma_f32_16x16x32_bf16(false, aF[kc], false, bF,
                                                      (short)0, acc, false, false);
      }
      const int cN = nt * 16 + colB;
      const float bb = sB1[cN], sc = sS2[cN], sh = sH2[cN];
#pragma unroll
      for (int v = 0; v < 8; ++v) {
        const int row = wave * 16 + v + rhi;
        const float g = gelu_fast(acc[v] + bb);
        sActB[row * DIMS + cN] = (bf16_t)fmaf(g, sc, sh);
      }
    }

    // ---- stage 2: @W2 + b2, gelu, * edge weight, atomic scatter ----
#pragma unroll
    for (int kc = 0; kc < 4; ++kc)
      aF[kc] = load_fragA(&sActB[rowA * DIMS + kc * 32 + kbA]);
#pragma unroll
    for (int nt = 0; nt < 8; ++nt) {
      v8f acc = {0.f, 0.f, 0.f, 0.f, 0.f, 0.f, 0.f, 0.f};
#pragma unroll
      for (int kc = 0; kc < 4; ++kc) {
        v16bf bF = load_fragB(&sWt2[(nt * 16 + colB) * DIMS + kc * 32 + kbB]);
        acc = __builtin_amdgcn_wmma_f32_16x16x32_bf16(false, aF[kc], false, bF,
                                                      (short)0, acc, false, false);
      }
      const int cN = nt * 16 + colB;
      const float bb = sB2[cN];
#pragma unroll
      for (int v = 0; v < 8; ++v) {
        const int row = wave * 16 + v + rhi;
        const float g = gelu_fast(acc[v] + bb) * sEw[row];
        atomicAdd(&agg[(size_t)sDst[row] * DIMS + cN], g);
      }
    }
  }
}

// ---------------------------------------------------------------------------
// Fused node update: concat(x, agg/cnt) -> BN -> @U1(256x128)+b -> gelu
//                    -> BN -> @U2(128x128)+b -> gelu -> out
// One 128-node tile per block.
// ---------------------------------------------------------------------------
__global__ __launch_bounds__(256) void node_update_kernel(
    const float* __restrict__ nodeF, const float* __restrict__ agg,
    const float* __restrict__ cnt,
    const float* __restrict__ u1g, const float* __restrict__ u1b,
    const float* __restrict__ u1m, const float* __restrict__ u1v,
    const float* __restrict__ u1W, const float* __restrict__ u1bias,
    const float* __restrict__ u2g, const float* __restrict__ u2b,
    const float* __restrict__ u2m, const float* __restrict__ u2v,
    const float* __restrict__ u2W, const float* __restrict__ u2bias,
    float* __restrict__ out) {
  __shared__ bf16_t sWt1[DIMS * 2 * DIMS];  // U1 transposed [n:128][k:256]
  __shared__ bf16_t sWt2[DIMS * DIMS];      // U2 transposed [n][k]
  __shared__ bf16_t sActA[128 * 256];
  __shared__ bf16_t sActB[128 * 128];
  __shared__ float sS1[256], sH1[256], sS2[DIMS], sH2[DIMS];
  __shared__ float sB1[DIMS], sB2[DIMS];

  const int tid  = threadIdx.x;
  const int wave = tid >> 5;
  const int lane = tid & 31;

  for (int i = tid; i < 256 * DIMS; i += 256) {
    const int k = i >> 7, n = i & (DIMS - 1);
    sWt1[n * 256 + k] = (bf16_t)u1W[i];
  }
  for (int i = tid; i < DIMS * DIMS; i += 256) {
    const int k = i >> 7, n = i & (DIMS - 1);
    sWt2[n * DIMS + k] = (bf16_t)u2W[i];
  }
  {
    const float s = u1g[tid] * rsqrtf(u1v[tid] + BN_EPS);
    sS1[tid] = s; sH1[tid] = u1b[tid] - u1m[tid] * s;
  }
  if (tid < DIMS) {
    const float s = u2g[tid] * rsqrtf(u2v[tid] + BN_EPS);
    sS2[tid] = s; sH2[tid] = u2b[tid] - u2m[tid] * s;
    sB1[tid] = u1bias[tid]; sB2[tid] = u2bias[tid];
  }
  __syncthreads();

  const int kbA  = (lane >> 4) << 3;
  const int kbB  = (lane >> 4) << 3;
  const int rowA = wave * 16 + (lane & 15);
  const int colB = lane & 15;
  const int rhi  = (lane >> 4) << 3;
  const int tile = blockIdx.x;

  // hoist per-lane BN u1 params (feature half + agg half)
  const int cf = lane * 4;
  const int ca = 128 + cf;
  float s1f[4], h1f[4], s1a[4], h1a[4];
#pragma unroll
  for (int j = 0; j < 4; ++j) {
    s1f[j] = sS1[cf + j]; h1f[j] = sH1[cf + j];
    s1a[j] = sS1[ca + j]; h1a[j] = sH1[ca + j];
  }

  // ---- gather concat(x, agg/cnt) + BN u1 -> bf16 [128 x 256] ----
#pragma unroll 4
  for (int r = 0; r < 16; ++r) {
    const int le = wave * 16 + r;
    const int i  = tile * 128 + le;
    float4 f = {0.f, 0.f, 0.f, 0.f}, a4 = {0.f, 0.f, 0.f, 0.f};
    float inv = 0.f;
    if (i < NNODES) {
      f   = *(const float4*)(nodeF + (size_t)i * DIMS + cf);
      a4  = *(const float4*)(agg + (size_t)i * DIMS + cf);
      inv = 1.0f / fmaxf(cnt[i], 1.0f);
    }
    bf16_t* d = &sActA[le * 256 + cf];
    d[0] = (bf16_t)fmaf(f.x, s1f[0], h1f[0]);
    d[1] = (bf16_t)fmaf(f.y, s1f[1], h1f[1]);
    d[2] = (bf16_t)fmaf(f.z, s1f[2], h1f[2]);
    d[3] = (bf16_t)fmaf(f.w, s1f[3], h1f[3]);
    bf16_t* d2 = &sActA[le * 256 + ca];
    d2[0] = (bf16_t)fmaf(a4.x * inv, s1a[0], h1a[0]);
    d2[1] = (bf16_t)fmaf(a4.y * inv, s1a[1], h1a[1]);
    d2[2] = (bf16_t)fmaf(a4.z * inv, s1a[2], h1a[2]);
    d2[3] = (bf16_t)fmaf(a4.w * inv, s1a[3], h1a[3]);
  }

  // ---- stage 1: K=256 GEMM, gelu, fold BN u2 -> sActB ----
  {
    v16bf aF[8];
#pragma unroll
    for (int kc = 0; kc < 8; ++kc)
      aF[kc] = load_fragA(&sActA[rowA * 256 + kc * 32 + kbA]);
#pragma unroll
    for (int nt = 0; nt < 8; ++nt) {
      v8f acc = {0.f, 0.f, 0.f, 0.f, 0.f, 0.f, 0.f, 0.f};
#pragma unroll
      for (int kc = 0; kc < 8; ++kc) {
        v16bf bF = load_fragB(&sWt1[(nt * 16 + colB) * 256 + kc * 32 + kbB]);
        acc = __builtin_amdgcn_wmma_f32_16x16x32_bf16(false, aF[kc], false, bF,
                                                      (short)0, acc, false, false);
      }
      const int cN = nt * 16 + colB;
      const float bb = sB1[cN], sc = sS2[cN], sh = sH2[cN];
#pragma unroll
      for (int v = 0; v < 8; ++v) {
        const int row = wave * 16 + v + rhi;
        const float g = gelu_fast(acc[v] + bb);
        sActB[row * DIMS + cN] = (bf16_t)fmaf(g, sc, sh);
      }
    }
  }

  // ---- stage 2: K=128 GEMM, gelu -> out ----
  {
    v16bf aF[4];
#pragma unroll
    for (int kc = 0; kc < 4; ++kc)
      aF[kc] = load_fragA(&sActB[rowA * DIMS + kc * 32 + kbA]);
#pragma unroll
    for (int nt = 0; nt < 8; ++nt) {
      v8f acc = {0.f, 0.f, 0.f, 0.f, 0.f, 0.f, 0.f, 0.f};
#pragma unroll
      for (int kc = 0; kc < 4; ++kc) {
        v16bf bF = load_fragB(&sWt2[(nt * 16 + colB) * DIMS + kc * 32 + kbB]);
        acc = __builtin_amdgcn_wmma_f32_16x16x32_bf16(false, aF[kc], false, bF,
                                                      (short)0, acc, false, false);
      }
      const int cN = nt * 16 + colB;
      const float bb = sB2[cN];
#pragma unroll
      for (int v = 0; v < 8; ++v) {
        const int row = wave * 16 + v + rhi;
        const int i = tile * 128 + row;
        if (i < NNODES) out[(size_t)i * DIMS + cN] = gelu_fast(acc[v] + bb);
      }
    }
  }
}

// ---------------------------------------------------------------------------
extern "C" void kernel_launch(void* const* d_in, const int* in_sizes, int n_in,
                              void* d_out, int out_size, void* d_ws, size_t ws_size,
                              hipStream_t stream) {
  const float* nodeF  = (const float*)d_in[0];
  const int*   edges  = (const int*)d_in[1];
  const float* edgeW  = (const float*)d_in[2];
  const float *p1g = (const float*)d_in[3],  *p1b = (const float*)d_in[4];
  const float *p1m = (const float*)d_in[5],  *p1v = (const float*)d_in[6];
  const float *p1W = (const float*)d_in[7],  *p1bias = (const float*)d_in[8];
  const float *p2g = (const float*)d_in[9],  *p2b = (const float*)d_in[10];
  const float *p2m = (const float*)d_in[11], *p2v = (const float*)d_in[12];
  const float *p2W = (const float*)d_in[13], *p2bias = (const float*)d_in[14];
  const float *u1g = (const float*)d_in[15], *u1b = (const float*)d_in[16];
  const float *u1m = (const float*)d_in[17], *u1v = (const float*)d_in[18];
  const float *u1W = (const float*)d_in[19], *u1bias = (const float*)d_in[20];
  const float *u2g = (const float*)d_in[21], *u2b = (const float*)d_in[22];
  const float *u2m = (const float*)d_in[23], *u2v = (const float*)d_in[24];
  const float *u2W = (const float*)d_in[25], *u2bias = (const float*)d_in[26];
  float* out = (float*)d_out;

  float* agg = (float*)d_ws;                       // [NNODES x 128] f32
  float* cnt = agg + (size_t)NNODES * DIMS;        // [NNODES] f32

  zero_kernel<<<512, 256, 0, stream>>>(agg, cnt);
  count_kernel<<<640, 256, 0, stream>>>(edges, cnt);
  edge_msg_kernel<<<1280, 256, 0, stream>>>(
      nodeF, edges, edgeW,
      p1g, p1b, p1m, p1v, p1W, p1bias,
      p2g, p2b, p2m, p2v, p2W, p2bias, agg);
  node_update_kernel<<<(NNODES + 127) / 128, 256, 0, stream>>>(
      nodeF, agg, cnt,
      u1g, u1b, u1m, u1v, u1W, u1bias,
      u2g, u2b, u2m, u2v, u2W, u2bias, out);
}
